// DGCNN_9414568313303
// MI455X (gfx1250) — matrix operand order repaired and checked by reading, not compiled, not run on hardware
//
#include <hip/hip_runtime.h>

#define BATCH 8
#define NPTS  2048
#define KNBR  20

typedef __attribute__((ext_vector_type(16))) __bf16 v16bf;
typedef __attribute__((ext_vector_type(8)))  float  v8f;

// ---------- bf16 helpers (RNE) ----------
__device__ __forceinline__ unsigned short f2bf(float f) {
    unsigned int u = __float_as_uint(f);
    u += 0x7FFFu + ((u >> 16) & 1u);
    return (unsigned short)(u >> 16);
}
__device__ __forceinline__ float bf2f(unsigned short h) {
    return __uint_as_float(((unsigned int)h) << 16);
}

// ---------- WMMA fragment loaders (CDNA5 §7.12.2 layouts) ----------
// A: 16x32 bf16. lane L: m=L&15, half=L>>4.
// VGPR v holds k = (v<4 ? 2v : 16+2(v-4)) + 8*half , +1
__device__ __forceinline__ v16bf load_A(const unsigned short* s, int lane, int ld) {
    int m = lane & 15, half = lane >> 4;
    v16bf a;
#pragma unroll
    for (int v = 0; v < 8; ++v) {
        int kb = ((v < 4) ? 2 * v : 16 + 2 * (v - 4)) + half * 8;
        a[2 * v]     = __builtin_bit_cast(__bf16, s[m * ld + kb]);
        a[2 * v + 1] = __builtin_bit_cast(__bf16, s[m * ld + kb + 1]);
    }
    return a;
}
// B: 32x16 bf16, B[k][n] = W[n][k] (row n of source, stride ld).
// element j: k = (j<8 ? j : 16+(j-8)) + 8*half ; n = lane&15.
__device__ __forceinline__ v16bf load_B(const unsigned short* w, int lane, int ld) {
    int n = lane & 15, half = lane >> 4;
    const unsigned short* p = w + (size_t)n * ld + half * 8;
    v16bf b;
#pragma unroll
    for (int j = 0; j < 16; ++j) b[j] = __builtin_bit_cast(__bf16, p[j < 8 ? j : j + 8]);
    return b;
}

__device__ __forceinline__ void atomicMaxF(float* addr, float val) {
    unsigned int* ua = (unsigned int*)addr;
    unsigned int old = *ua;
    while (true) {
        if (__uint_as_float(old) >= val) break;
        unsigned int assumed = old;
        old = atomicCAS(ua, assumed, __float_as_uint(val));
        if (old == assumed) break;
    }
}

// sorted-descending top-K insertion; ties keep earlier index (jax top_k order)
__device__ __forceinline__ void topk_insert(float v, int id, float* vals, int* ids) {
    if (v <= vals[KNBR - 1]) return;
    int p = KNBR - 1;
    while (p > 0 && vals[p - 1] < v) {
        vals[p] = vals[p - 1];
        ids[p]  = ids[p - 1];
        --p;
    }
    vals[p] = v;
    ids[p]  = id;
}

// ---------- small utility kernels ----------
__global__ void k_init_out(float* out, int n) {
    int i = blockIdx.x * blockDim.x + threadIdx.x;
    if (i < n) out[i] = -INFINITY;
}
__global__ void k_convert_w(const float* __restrict__ w, unsigned short* __restrict__ dst,
                            int O, int Cs, int Kp) {
    int i = blockIdx.x * blockDim.x + threadIdx.x;
    if (i >= O * Kp) return;
    int o = i / Kp, c = i - o * Kp;
    dst[i] = (c < Cs) ? f2bf(w[o * Cs + c]) : (unsigned short)0;
}
__global__ void k_convert_x(const float* __restrict__ x, unsigned short* __restrict__ dst, int n) {
    int i = blockIdx.x * blockDim.x + threadIdx.x;
    if (i < n) dst[i] = f2bf(x[i]);
}
__global__ void k_xx(const unsigned short* __restrict__ x, float* __restrict__ xx, int C) {
    int i = blockIdx.x * blockDim.x + threadIdx.x;
    if (i >= BATCH * NPTS) return;
    float s = 0.f;
    for (int c = 0; c < C; ++c) {
        float v = bf2f(x[(size_t)i * C + c]);
        s += v * v;
    }
    xx[i] = s;
}
__global__ void k_finalize(const float* __restrict__ stats, const float* __restrict__ g,
                           const float* __restrict__ be, float* __restrict__ stats2,
                           int O, float cnt) {
    int o = threadIdx.x;
    if (o >= O) return;
    float s = 0.f, q = 0.f;
    for (int sl = 0; sl < 64; ++sl) {
        s += stats[sl * 512 + o * 2];
        q += stats[sl * 512 + o * 2 + 1];
    }
    float mean  = s / cnt;
    float var   = q / cnt - mean * mean;
    float scale = g[o] * rsqrtf(var + 1e-5f);
    stats2[o * 2]     = scale;
    stats2[o * 2 + 1] = be[o] - mean * scale;
}

// ---------- EdgeConv: one block (1 wave) per point; 20 rows padded to 32 ----------
// mode 0: accumulate BN stats.  mode 1: apply BN + LeakyReLU + max over k, write bf16.
__global__ __launch_bounds__(32) void k_edge(
    const unsigned short* __restrict__ xin, const int* __restrict__ idx,
    const unsigned short* __restrict__ wbf, float* __restrict__ stats,
    const float* __restrict__ stats2, unsigned short* __restrict__ xout,
    unsigned short* __restrict__ catbf, int C, int O, int k2sh, int cat_off, int mode) {
    __shared__ unsigned short eLDS[32 * 256];  // 32 rows x K2p(<=256) bf16
    const int K2p  = 1 << k2sh;
    const int lane = threadIdx.x;
    const int gid  = blockIdx.x;  // b*NPTS + n
    const int b    = gid >> 11;
    const int n    = gid & (NPTS - 1);
    const unsigned short* xb = xin + (size_t)b * NPTS * C;

    // build edge-feature tile e = [nbr-ctr | ctr | 0-pad], rows>=20 are zero
    const int total = 32 << k2sh;
    for (int t = lane; t < total; t += 32) {
        int r = t >> k2sh;
        int c = t & (K2p - 1);
        float v = 0.f;
        if (r < KNBR) {
            if (c < C) {
                int nb = idx[(size_t)gid * KNBR + r];
                v = bf2f(xb[(size_t)nb * C + c]) - bf2f(xb[(size_t)n * C + c]);
            } else if (c < 2 * C) {
                v = bf2f(xb[(size_t)n * C + (c - C)]);
            }
        }
        eLDS[t] = f2bf(v);
    }
    __syncthreads();

    const int half = lane >> 4, col = lane & 15;
    const int nOT = O >> 4, nKT = K2p >> 5;
    for (int ot = 0; ot < nOT; ++ot) {
        int o0 = ot * 16;
        __builtin_prefetch(wbf + (size_t)o0 * K2p, 0, 3);
        v8f acc0 = {0, 0, 0, 0, 0, 0, 0, 0};
        v8f acc1 = {0, 0, 0, 0, 0, 0, 0, 0};
        for (int kt = 0; kt < nKT; ++kt) {
            v16bf a0 = load_A(eLDS + kt * 32, lane, K2p);
            v16bf a1 = load_A(eLDS + 16 * K2p + kt * 32, lane, K2p);
            v16bf bb = load_B(wbf + (size_t)o0 * K2p + kt * 32, lane, K2p);
            acc0 = __builtin_amdgcn_wmma_f32_16x16x32_bf16(false, a0, false, bb, (short)0, acc0, false, false);
            acc1 = __builtin_amdgcn_wmma_f32_16x16x32_bf16(false, a1, false, bb, (short)0, acc1, false, false);
        }
        int o = o0 + col;
        if (mode == 0) {
            float s = 0.f, q = 0.f;
#pragma unroll
            for (int r = 0; r < 8; ++r) { float v = acc0[r]; s += v; q += v * v; }   // kk 0..15
            if (half == 0) {
#pragma unroll
                for (int r = 0; r < 4; ++r) { float v = acc1[r]; s += v; q += v * v; } // kk 16..19
            }
            s += __shfl_xor(s, 16, 32);
            q += __shfl_xor(q, 16, 32);
            if (lane < 16) {
                float* st = stats + (size_t)(gid & 63) * 512;  // 64-way sliced
                atomicAdd(&st[o * 2 + 0], s);
                atomicAdd(&st[o * 2 + 1], q);
            }
        } else {
            float scale = stats2[o * 2], shift = stats2[o * 2 + 1];
            float mx = -INFINITY;
#pragma unroll
            for (int r = 0; r < 8; ++r) {
                float v = acc0[r] * scale + shift;
                v = v >= 0.f ? v : 0.2f * v;
                mx = fmaxf(mx, v);
            }
            if (half == 0) {
#pragma unroll
                for (int r = 0; r < 4; ++r) {
                    float v = acc1[r] * scale + shift;
                    v = v >= 0.f ? v : 0.2f * v;
                    mx = fmaxf(mx, v);
                }
            }
            mx = fmaxf(mx, __shfl_xor(mx, 16, 32));
            if (lane < 16) {
                unsigned short hv = f2bf(mx);
                xout[(size_t)gid * O + o]               = hv;
                catbf[(size_t)gid * 512 + cat_off + o]  = hv;
            }
        }
    }
}

// ---------- kNN: block = 16 rows of one batch; dist strip lives in LDS ----------
__global__ __launch_bounds__(128) void k_knn(const unsigned short* __restrict__ x,
                                             const float* __restrict__ xx,
                                             int* __restrict__ outidx, int csh) {
    extern __shared__ char smem[];
    const int C = 1 << csh;
    float*          negd  = (float*)smem;                               // 16*2048 f32
    unsigned short* rowbf = (unsigned short*)(negd + 16 * NPTS);        // 16*256 bf16
    float*          xxm   = (float*)(rowbf + 16 * 256);                 // 16
    float*          mval  = xxm + 16;                                   // 16*8*20
    int*            midx  = (int*)(mval + 16 * 8 * KNBR);               // 16*8*20

    int blk = blockIdx.x;
    int b = blk >> 7, rt = blk & 127;
    int rowbase = rt * 16;
    const unsigned short* xb = x + (size_t)b * NPTS * C;

    for (int t = threadIdx.x; t < 16 * C; t += 128) {
        int r = t >> csh, c = t & (C - 1);
        rowbf[t] = xb[(size_t)(rowbase + r) * C + c];
    }
    if (threadIdx.x < 16) xxm[threadIdx.x] = xx[b * NPTS + rowbase + threadIdx.x];
    __syncthreads();

    int wv = threadIdx.x >> 5, lane = threadIdx.x & 31;
    int half = lane >> 4, col = lane & 15;
    for (int ct = wv; ct < NPTS / 16; ct += 4) {
        int colbase = ct * 16;
        v8f acc = {0, 0, 0, 0, 0, 0, 0, 0};
        for (int kt = 0; kt < (C >> 5); ++kt) {
            v16bf a  = load_A(rowbf + kt * 32, lane, C);
            v16bf bb = load_B(xb + (size_t)colbase * C + kt * 32, lane, C);
            acc = __builtin_amdgcn_wmma_f32_16x16x32_bf16(false, a, false, bb, (short)0, acc, false, false);
        }
        int nc = colbase + col;
        float xxn = xx[b * NPTS + nc];
#pragma unroll
        for (int r = 0; r < 8; ++r) {
            int m = r + 8 * half;
            negd[m * NPTS + nc] = 2.f * acc[r] - xxm[m] - xxn;  // neg squared dist
        }
    }
    __syncthreads();

    {   // per-segment top-20 (8 threads per row, 256 cols each)
        int r = threadIdx.x >> 3, seg = threadIdx.x & 7;
        float tv[KNBR]; int ti[KNBR];
        for (int j = 0; j < KNBR; ++j) { tv[j] = -INFINITY; ti[j] = 0; }
        const float* row = negd + r * NPTS;
        int c0 = seg * 256;
        for (int c = c0; c < c0 + 256; ++c) topk_insert(row[c], c, tv, ti);
        float* mv = mval + (r * 8 + seg) * KNBR;
        int*   mi = midx + (r * 8 + seg) * KNBR;
        for (int j = 0; j < KNBR; ++j) { mv[j] = tv[j]; mi[j] = ti[j]; }
    }
    __syncthreads();
    if (threadIdx.x < 16) {  // merge 8 sorted lists per row
        int r = threadIdx.x;
        float tv[KNBR]; int ti[KNBR];
        for (int j = 0; j < KNBR; ++j) { tv[j] = -INFINITY; ti[j] = 0; }
        for (int seg = 0; seg < 8; ++seg) {
            const float* mv = mval + (r * 8 + seg) * KNBR;
            const int*   mi = midx + (r * 8 + seg) * KNBR;
            for (int j = 0; j < KNBR; ++j) topk_insert(mv[j], mi[j], tv, ti);
        }
        int* out = outidx + (size_t)(b * NPTS + rowbase + r) * KNBR;
        for (int j = 0; j < KNBR; ++j) out[j] = ti[j];
    }
}

// ---------- final 1x1 conv [512->1024] + global max over N ----------
__global__ __launch_bounds__(32) void k_final(const unsigned short* __restrict__ catbf,
                                              const unsigned short* __restrict__ wfbf,
                                              const float* __restrict__ bfp,
                                              float* __restrict__ out) {
    int blk = blockIdx.x;                 // B * 64 otiles * 8 nchunks
    int b = blk >> 9;
    int rem = blk & 511;
    int ot = rem >> 3, nchunk = rem & 7;
    int lane = threadIdx.x, col = lane & 15;
    int o0 = ot * 16;
    const unsigned short* wb = wfbf + (size_t)o0 * 512;
    float mx = -INFINITY;
    for (int rtile = 0; rtile < 16; ++rtile) {
        int row0 = nchunk * 256 + rtile * 16;
        const unsigned short* ab = catbf + ((size_t)b * NPTS + row0) * 512;
        v8f acc = {0, 0, 0, 0, 0, 0, 0, 0};
        for (int kt = 0; kt < 16; ++kt) {
            v16bf a  = load_A(ab + kt * 32, lane, 512);
            v16bf bb = load_B(wb + kt * 32, lane, 512);
            acc = __builtin_amdgcn_wmma_f32_16x16x32_bf16(false, a, false, bb, (short)0, acc, false, false);
        }
#pragma unroll
        for (int r = 0; r < 8; ++r) mx = fmaxf(mx, acc[r]);
    }
    mx = fmaxf(mx, __shfl_xor(mx, 16, 32));
    if (lane < 16) {
        int o = o0 + col;
        atomicMaxF(&out[b * 1024 + o], mx + bfp[o]);
    }
}

// ---------- host orchestration ----------
extern "C" void kernel_launch(void* const* d_in, const int* in_sizes, int n_in,
                              void* d_out, int out_size, void* d_ws, size_t ws_size,
                              hipStream_t stream) {
    (void)in_sizes; (void)n_in; (void)out_size; (void)ws_size;
    const float* x    = (const float*)d_in[0];
    const int*   idx0 = (const int*)d_in[1];
    const float* w[4]  = {(const float*)d_in[2], (const float*)d_in[5],
                          (const float*)d_in[8], (const float*)d_in[11]};
    const float* g[4]  = {(const float*)d_in[3], (const float*)d_in[6],
                          (const float*)d_in[9], (const float*)d_in[12]};
    const float* be[4] = {(const float*)d_in[4], (const float*)d_in[7],
                          (const float*)d_in[10], (const float*)d_in[13]};
    const float* wf  = (const float*)d_in[14];
    const float* bfp = (const float*)d_in[15];
    float* out = (float*)d_out;
    char* ws = (char*)d_ws;

    // workspace layout (all 256-aligned), ~36.3 MB total
    const size_t OFF_WBF0  = 0;                       // 64*32  bf16
    const size_t OFF_WBF1  = OFF_WBF0 + 64 * 32 * 2;  // 64*128
    const size_t OFF_WBF2  = OFF_WBF1 + 64 * 128 * 2; // 128*128
    const size_t OFF_WBF3  = OFF_WBF2 + 128 * 128 * 2;// 256*256
    const size_t OFF_WFBF  = OFF_WBF3 + 256 * 256 * 2;// 1024*512
    const size_t OFF_XA    = OFF_WFBF + 1024 * 512 * 2;
    const size_t OFF_XB    = OFF_XA + (size_t)BATCH * NPTS * 256 * 2;
    const size_t OFF_CAT   = OFF_XB + (size_t)BATCH * NPTS * 256 * 2;
    const size_t OFF_IDX   = OFF_CAT + (size_t)BATCH * NPTS * 512 * 2;
    const size_t OFF_XX    = OFF_IDX + (size_t)BATCH * NPTS * KNBR * 4;
    const size_t OFF_STATS = OFF_XX + (size_t)BATCH * NPTS * 4;
    const size_t OFF_ST2   = OFF_STATS + 64 * 512 * 4;

    unsigned short* wbf[4] = {(unsigned short*)(ws + OFF_WBF0), (unsigned short*)(ws + OFF_WBF1),
                              (unsigned short*)(ws + OFF_WBF2), (unsigned short*)(ws + OFF_WBF3)};
    unsigned short* wfbf  = (unsigned short*)(ws + OFF_WFBF);
    unsigned short* xA    = (unsigned short*)(ws + OFF_XA);
    unsigned short* xB    = (unsigned short*)(ws + OFF_XB);
    unsigned short* catbf = (unsigned short*)(ws + OFF_CAT);
    int*   idxbuf = (int*)(ws + OFF_IDX);
    float* xxbuf  = (float*)(ws + OFF_XX);
    float* stats  = (float*)(ws + OFF_STATS);
    float* stats2 = (float*)(ws + OFF_ST2);

    const int Os[4]  = {64, 64, 128, 256};
    const int Cs[4]  = {6, 128, 128, 256};   // source 2C widths of weights
    const int Kp[4]  = {32, 128, 128, 256};  // padded K dims

    k_init_out<<<(BATCH * 1024 + 255) / 256, 256, 0, stream>>>(out, BATCH * 1024);
    for (int i = 0; i < 4; ++i) {
        int n = Os[i] * Kp[i];
        k_convert_w<<<(n + 255) / 256, 256, 0, stream>>>(w[i], wbf[i], Os[i], Cs[i], Kp[i]);
    }
    k_convert_w<<<(1024 * 512 + 255) / 256, 256, 0, stream>>>(wf, wfbf, 1024, 512, 512);
    k_convert_x<<<(BATCH * NPTS * 3 + 255) / 256, 256, 0, stream>>>(x, xA, BATCH * NPTS * 3);

    const float cnt = (float)(BATCH * NPTS * KNBR);
    const size_t KNN_SMEM = 16 * NPTS * 4 + 16 * 256 * 2 + 16 * 4
                          + 16 * 8 * KNBR * 4 + 16 * 8 * KNBR * 4;

    auto edge = [&](const unsigned short* xin, const int* idxp, int li,
                    unsigned short* xout, int C, int k2sh, int cat_off) {
        hipMemsetAsync(stats, 0, 64 * 512 * 4, stream);
        k_edge<<<BATCH * NPTS, 32, 0, stream>>>(xin, idxp, wbf[li], stats, nullptr,
                                                nullptr, nullptr, C, Os[li], k2sh, cat_off, 0);
        k_finalize<<<1, 256, 0, stream>>>(stats, g[li], be[li], stats2, Os[li], cnt);
        k_edge<<<BATCH * NPTS, 32, 0, stream>>>(xin, idxp, wbf[li], stats, stats2,
                                                xout, catbf, C, Os[li], k2sh, cat_off, 1);
    };
    auto knn = [&](const unsigned short* xin, int C, int csh) {
        k_xx<<<(BATCH * NPTS + 255) / 256, 256, 0, stream>>>(xin, xxbuf, C);
        k_knn<<<BATCH * (NPTS / 16), 128, KNN_SMEM, stream>>>(xin, xxbuf, idxbuf, csh);
    };

    // layer 0: given indices, C=3 (K2p=32), O=64
    edge(xA, idx0, 0, xB, 3, 5, 0);
    // layer 1: knn on 64-dim feats, O=64
    knn(xB, 64, 6);
    edge(xB, idxbuf, 1, xA, 64, 7, 64);
    // layer 2: knn on 64-dim feats, O=128
    knn(xA, 64, 6);
    edge(xA, idxbuf, 2, xB, 64, 7, 128);
    // layer 3: knn on 128-dim feats, O=256
    knn(xB, 128, 7);
    edge(xB, idxbuf, 3, xA, 128, 8, 256);
    // final conv + global max pool
    k_final<<<BATCH * 64 * 8, 32, 0, stream>>>(catbf, wfbf, bfp, out);
}
